// GCNMasker_21912923144344
// MI455X (gfx1250) — compile-verified, matrix operand order checked by reading, not run on hardware
//
#include <hip/hip_runtime.h>
#include <hip/hip_bf16.h>
#include <math.h>

typedef __attribute__((ext_vector_type(16))) _Float16 v16h;
typedef __attribute__((ext_vector_type(8)))  float    v8f;
typedef __attribute__((ext_vector_type(4)))  int      v4i;
typedef __attribute__((address_space(1))) v4i* gv4i_p;   // global AS pointer
typedef __attribute__((address_space(3))) v4i* lv4i_p;   // LDS AS pointer

#define D_H 128

#if __has_builtin(__builtin_amdgcn_global_load_async_to_lds_b128) && \
    __has_builtin(__builtin_amdgcn_s_wait_asynccnt)
#define HAVE_ASYNC_LDS 1
#else
#define HAVE_ASYNC_LDS 0
#endif

// ---------------------------------------------------------------- utilities
__global__ void zero_f32(float* p, long n) {
    long i = (long)blockIdx.x * blockDim.x + threadIdx.x;
    if (i < n) p[i] = 0.0f;
}

__global__ void deg_kernel(const int* __restrict__ row, const int* __restrict__ col,
                           float* __restrict__ deg_out, float* __restrict__ deg_in, int E) {
    int e = blockIdx.x * blockDim.x + threadIdx.x;
    if (e < E) {
        atomicAdd(deg_out + row[e], 1.0f);
        atomicAdd(deg_in  + col[e], 1.0f);
    }
}

__global__ void rsqrt_kernel(float* p, int n) {
    int i = blockIdx.x * blockDim.x + threadIdx.x;
    if (i < n) p[i] = rsqrtf(fmaxf(p[i], 1.0f));
}

__global__ void f32_to_f16(const float* __restrict__ src, _Float16* __restrict__ dst, long n) {
    long i = (long)blockIdx.x * blockDim.x + threadIdx.x;
    if (i < n) dst[i] = (_Float16)src[i];
}

// Pack W[K,Nc] (f32, row-major) into WMMA B-fragment order (f16).
// B fragment: lane L holds B[kc*32 + (L>>4)*16 + i][nt*16 + (L&15)], i=0..15 contiguous.
__global__ void pack_wmma_b(const float* __restrict__ W, _Float16* __restrict__ Wp,
                            int K, int Nc) {
    int gid = blockIdx.x * blockDim.x + threadIdx.x;
    int total = K * Nc;
    if (gid >= total) return;
    int k = gid / Nc, n = gid % Nc;
    int kc = k >> 5, kl = k & 31;
    int hi = kl >> 4, i = kl & 15;
    int nt = n >> 4, nl = n & 15;
    int lane = (hi << 4) | nl;
    int NT = Nc >> 4;
    size_t dst = ((size_t)(kc * NT + nt) * 32 + lane) * 16 + i;
    Wp[dst] = (_Float16)W[(size_t)k * Nc + n];
}

// GCN message scatter: agg[col[e]] += h[row[e]] * ns[row[e]]  (per feature dim)
__global__ void scatter_kernel(const float* __restrict__ h, const float* __restrict__ ns,
                               const int* __restrict__ row, const int* __restrict__ col,
                               float* __restrict__ agg, long total) {
    long gid = (long)blockIdx.x * blockDim.x + threadIdx.x;
    if (gid >= total) return;
    int e = (int)(gid >> 7);
    int d = (int)(gid & 127);
    int r = row[e], c = col[e];
    float v = h[(size_t)r * D_H + d] * ns[r];
    atomicAdd(agg + (size_t)c * D_H + d, v);
}

// ------------------------------------------------- dense 128x128 WMMA GEMM
// out[m,:] = opt_residual + opt_relu( (A[m,:]*opt_scale[m]) @ W + bias )
// One wave per 16-row tile; 4 K-chunks x 8 N-tiles = 32 v_wmma per tile.
__global__ __launch_bounds__(128)
void dense128_wmma(const float* __restrict__ A, const float* __restrict__ scale,
                   const _Float16* __restrict__ Wp, const float* __restrict__ bias,
                   const float* __restrict__ residual, int do_relu, int nrows,
                   float* __restrict__ out) {
    const int lane = threadIdx.x & 31;
    const int wv   = threadIdx.x >> 5;
    const int tile = blockIdx.x * 4 + wv;
    const int row0 = tile * 16;
    if (row0 >= nrows) return;                 // wave-uniform: EXEC stays all-ones
    const int nl = lane & 15;
    const int hi = lane >> 4;
    int arow = row0 + nl; if (arow >= nrows) arow = nrows - 1;
    const float s = scale ? scale[arow] : 1.0f;
    const float* arp = A + (size_t)arow * D_H;

    v8f acc[8] = {};
    #pragma unroll
    for (int kc = 0; kc < 4; ++kc) {
        const int koff = kc * 32 + hi * 16;
        v16h a;
        const float4* f4 = (const float4*)(arp + koff);
        #pragma unroll
        for (int q = 0; q < 4; ++q) {
            float4 v = f4[q];
            a[q*4+0] = (_Float16)(v.x * s);
            a[q*4+1] = (_Float16)(v.y * s);
            a[q*4+2] = (_Float16)(v.z * s);
            a[q*4+3] = (_Float16)(v.w * s);
        }
        #pragma unroll
        for (int nt = 0; nt < 8; ++nt) {
            v16h b = *(const v16h*)(Wp + ((size_t)(kc * 8 + nt) * 32 + lane) * 16);
            acc[nt] = __builtin_amdgcn_wmma_f32_16x16x32_f16(
                false, a, false, b, (short)0, acc[nt], false, false);
        }
    }
    #pragma unroll
    for (int nt = 0; nt < 8; ++nt) {
        const int c = nt * 16 + nl;
        const float bv = bias[c];
        #pragma unroll
        for (int r = 0; r < 8; ++r) {
            int m = row0 + r + hi * 8;
            if (m < nrows) {
                float v = acc[nt][r] + bv;
                if (do_relu) v = fmaxf(v, 0.0f);
                if (residual) v += residual[(size_t)m * D_H + c];
                out[(size_t)m * D_H + c] = v;
            }
        }
    }
}

// ------------------------------------------------------------ edge link MLP
// x = [h16[row[e]], h16[col[e]]]  (256)  ->128 relu ->64 relu ->1 sigmoid
// 256 threads = 8 waves; each wave owns 16 edges. Both packed weight matrices
// are staged into LDS once per block (async-to-LDS when available), cutting
// L2 weight traffic 8x. Layer0 A-fragments gathered straight from global
// (contiguous 32B per lane). Layer0 out -> LDS (f16) -> layer1 A-fragments.
__global__ __launch_bounds__(256)
void edge_mlp_wmma(const _Float16* __restrict__ h16,
                   const int* __restrict__ row, const int* __restrict__ col,
                   const _Float16* __restrict__ Wp0, const float* __restrict__ b0,
                   const _Float16* __restrict__ Wp1, const float* __restrict__ b1,
                   const float* __restrict__ W2, const float* __restrict__ b2,
                   float* __restrict__ link, int E) {
    __shared__ __align__(32) _Float16 wlds0[256 * 128];     // 64 KB packed mlp_W0
    __shared__ __align__(32) _Float16 wlds1[128 * 64];      // 16 KB packed mlp_W1
    __shared__ __align__(32) _Float16 ylds[8][16][D_H];     // 32 KB layer-0 activations

    // ---- cooperative weight staging (once per block) ----
#if HAVE_ASYNC_LDS
    for (int i = threadIdx.x; i < 4096; i += 256)           // 4096 x 16B = 64 KB
        __builtin_amdgcn_global_load_async_to_lds_b128(
            (gv4i_p)(Wp0 + (size_t)i * 8),
            (lv4i_p)&wlds0[(size_t)i * 8], 0, 0);
    for (int i = threadIdx.x; i < 1024; i += 256)           // 1024 x 16B = 16 KB
        __builtin_amdgcn_global_load_async_to_lds_b128(
            (gv4i_p)(Wp1 + (size_t)i * 8),
            (lv4i_p)&wlds1[(size_t)i * 8], 0, 0);
    __builtin_amdgcn_s_wait_asynccnt(0);
#else
    for (int i = threadIdx.x; i < 4096; i += 256)
        ((uint4*)wlds0)[i] = ((const uint4*)Wp0)[i];
    for (int i = threadIdx.x; i < 1024; i += 256)
        ((uint4*)wlds1)[i] = ((const uint4*)Wp1)[i];
#endif
    __syncthreads();

    const int lane = threadIdx.x & 31;
    const int wv   = threadIdx.x >> 5;
    const int nl   = lane & 15;
    const int hi   = lane >> 4;
    int e0 = (blockIdx.x * 8 + wv) * 16;
    if (e0 + 16 > E) e0 = (E >= 16) ? (E - 16) : 0;   // wave-uniform clamp (dup work)
    const int e = e0 + nl;
    const _Float16* rowp = h16 + (size_t)row[e] * D_H;
    const _Float16* colp = h16 + (size_t)col[e] * D_H;

    // ---- layer 0: [16,256] @ [256,128]
    v8f acc[8] = {};
    #pragma unroll
    for (int kc = 0; kc < 8; ++kc) {
        const int koff = kc * 32 + hi * 16;
        const _Float16* src = (koff < 128) ? (rowp + koff) : (colp + (koff - 128));
        v16h a = *(const v16h*)src;
        #pragma unroll
        for (int nt = 0; nt < 8; ++nt) {
            v16h b = *(const v16h*)&wlds0[((size_t)(kc * 8 + nt) * 32 + lane) * 16];
            acc[nt] = __builtin_amdgcn_wmma_f32_16x16x32_f16(
                false, a, false, b, (short)0, acc[nt], false, false);
        }
    }
    #pragma unroll
    for (int nt = 0; nt < 8; ++nt) {
        const int c = nt * 16 + nl;
        const float bv = b0[c];
        #pragma unroll
        for (int r = 0; r < 8; ++r)
            ylds[wv][r + hi * 8][c] = (_Float16)fmaxf(acc[nt][r] + bv, 0.0f);
    }
    __syncthreads();

    // ---- layer 1: [16,128] @ [128,64]
    v8f acc1[4] = {};
    #pragma unroll
    for (int kc = 0; kc < 4; ++kc) {
        v16h a = *(const v16h*)&ylds[wv][nl][kc * 32 + hi * 16];
        #pragma unroll
        for (int nt = 0; nt < 4; ++nt) {
            v16h b = *(const v16h*)&wlds1[((size_t)(kc * 4 + nt) * 32 + lane) * 16];
            acc1[nt] = __builtin_amdgcn_wmma_f32_16x16x32_f16(
                false, a, false, b, (short)0, acc1[nt], false, false);
        }
    }

    // ---- layer 2: dot(relu(z), W2) + b2, sigmoid; reduce across 16 lanes
    float w2v[4], b1v[4];
    #pragma unroll
    for (int nt = 0; nt < 4; ++nt) {
        w2v[nt] = W2[nt * 16 + nl];
        b1v[nt] = b1[nt * 16 + nl];
    }
    const float b2v = b2[0];
    #pragma unroll
    for (int r = 0; r < 8; ++r) {
        float p = 0.0f;
        #pragma unroll
        for (int nt = 0; nt < 4; ++nt)
            p += fmaxf(acc1[nt][r] + b1v[nt], 0.0f) * w2v[nt];
        p += __shfl_xor(p, 1, 16);
        p += __shfl_xor(p, 2, 16);
        p += __shfl_xor(p, 4, 16);
        p += __shfl_xor(p, 8, 16);
        if (nl == 0) {
            int ee = e0 + r + hi * 8;
            link[ee] = 1.0f / (1.0f + __expf(-(p + b2v)));
        }
    }
}

// ------------------------------------------------------------- node MLP
// 128 -> 64 relu -> 32 relu -> 1 sigmoid (tiny: 0.2 GFLOP total, scalar ok)
__global__ __launch_bounds__(128)
void node_mlp(const float* __restrict__ h,
              const float* __restrict__ W0, const float* __restrict__ b0,
              const float* __restrict__ W1, const float* __restrict__ b1,
              const float* __restrict__ W2, const float* __restrict__ b2,
              float* __restrict__ out, int n) {
    int i = blockIdx.x * blockDim.x + threadIdx.x;
    if (i >= n) return;
    const float* x = h + (size_t)i * D_H;
    float x1[64];
    for (int j = 0; j < 64; ++j) {
        float s = b0[j];
        for (int k = 0; k < 128; ++k) s += x[k] * W0[k * 64 + j];
        x1[j] = fmaxf(s, 0.0f);
    }
    float x2[32];
    for (int j = 0; j < 32; ++j) {
        float s = b1[j];
        for (int k = 0; k < 64; ++k) s += x1[k] * W1[k * 32 + j];
        x2[j] = fmaxf(s, 0.0f);
    }
    float s = b2[0];
    for (int k = 0; k < 32; ++k) s += x2[k] * W2[k];
    out[i] = 1.0f / (1.0f + __expf(-s));
}

// ---------------------------------------------------------------- launcher
extern "C" void kernel_launch(void* const* d_in, const int* in_sizes, int n_in,
                              void* d_out, int out_size, void* d_ws, size_t ws_size,
                              hipStream_t stream) {
    const float* h_in   = (const float*)d_in[0];
    const int*   row    = (const int*)d_in[1];
    const int*   col    = (const int*)d_in[2];
    const float* emb_W  = (const float*)d_in[3];
    const float* emb_b  = (const float*)d_in[4];
    const float* gcn_W1 = (const float*)d_in[5];
    const float* gcn_b1 = (const float*)d_in[6];
    const float* gcn_W2 = (const float*)d_in[7];
    const float* gcn_b2 = (const float*)d_in[8];
    const float* mlp_W0 = (const float*)d_in[9];
    const float* mlp_b0 = (const float*)d_in[10];
    const float* mlp_W1 = (const float*)d_in[11];
    const float* mlp_b1 = (const float*)d_in[12];
    const float* mlp_W2 = (const float*)d_in[13];
    const float* mlp_b2 = (const float*)d_in[14];
    const float* nW0 = (const float*)d_in[15];
    const float* nb0 = (const float*)d_in[16];
    const float* nW1 = (const float*)d_in[17];
    const float* nb1 = (const float*)d_in[18];
    const float* nW2 = (const float*)d_in[19];
    const float* nb2 = (const float*)d_in[20];

    const int N = in_sizes[0] / D_H;
    const int E = in_sizes[1];

    // workspace carve-out (256B aligned)
    char* ws = (char*)d_ws;
    size_t off = 0;
    auto alloc = [&](size_t bytes) -> void* {
        void* p = ws + off;
        off = (off + bytes + 255) & ~(size_t)255;
        return p;
    };
    float*     ns    = (float*)alloc((size_t)N * 4);          // rsqrt(deg_out)
    float*     nd    = (float*)alloc((size_t)N * 4);          // rsqrt(deg_in)
    float*     hA    = (float*)alloc((size_t)N * D_H * 4);
    float*     hB    = (float*)alloc((size_t)N * D_H * 4);
    float*     agg   = (float*)alloc((size_t)N * D_H * 4);
    _Float16*  h16   = (_Float16*)alloc((size_t)N * D_H * 2);
    _Float16*  WpEmb = (_Float16*)alloc(128 * 128 * 2);
    _Float16*  WpG1  = (_Float16*)alloc(128 * 128 * 2);
    _Float16*  WpG2  = (_Float16*)alloc(128 * 128 * 2);
    _Float16*  WpM0  = (_Float16*)alloc(256 * 128 * 2);
    _Float16*  WpM1  = (_Float16*)alloc(128 * 64 * 2);
    (void)ws_size;

    float* link_out = (float*)d_out;          // [E]
    float* node_out = (float*)d_out + E;      // [N]

    const int B = 256;
    long nf = (long)N * D_H;

    // degrees -> rsqrt norms
    zero_f32<<<(N + B - 1) / B, B, 0, stream>>>(ns, N);
    zero_f32<<<(N + B - 1) / B, B, 0, stream>>>(nd, N);
    deg_kernel<<<(E + B - 1) / B, B, 0, stream>>>(row, col, ns, nd, E);
    rsqrt_kernel<<<(N + B - 1) / B, B, 0, stream>>>(ns, N);
    rsqrt_kernel<<<(N + B - 1) / B, B, 0, stream>>>(nd, N);

    // pack weights into WMMA B-fragment layout (f16)
    pack_wmma_b<<<(128 * 128 + B - 1) / B, B, 0, stream>>>(emb_W,  WpEmb, 128, 128);
    pack_wmma_b<<<(128 * 128 + B - 1) / B, B, 0, stream>>>(gcn_W1, WpG1,  128, 128);
    pack_wmma_b<<<(128 * 128 + B - 1) / B, B, 0, stream>>>(gcn_W2, WpG2,  128, 128);
    pack_wmma_b<<<(256 * 128 + B - 1) / B, B, 0, stream>>>(mlp_W0, WpM0,  256, 128);
    pack_wmma_b<<<(128 * 64  + B - 1) / B, B, 0, stream>>>(mlp_W1, WpM1,  128, 64);

    const int gDense = (N + 63) / 64;
    // embedding: hA = h_in @ emb_W + emb_b
    dense128_wmma<<<gDense, 128, 0, stream>>>(h_in, nullptr, WpEmb, emb_b,
                                              nullptr, 0, N, hA);
    // GCN layer 1: hB = relu((scatter(hA*ns)*nd) @ W1 + b1) + hA
    zero_f32<<<(int)((nf + B - 1) / B), B, 0, stream>>>(agg, nf);
    scatter_kernel<<<(int)(((long)E * D_H + B - 1) / B), B, 0, stream>>>(
        hA, ns, row, col, agg, (long)E * D_H);
    dense128_wmma<<<gDense, 128, 0, stream>>>(agg, nd, WpG1, gcn_b1, hA, 1, N, hB);
    // GCN layer 2: hA = relu((scatter(hB*ns)*nd) @ W2 + b2) + hB
    zero_f32<<<(int)((nf + B - 1) / B), B, 0, stream>>>(agg, nf);
    scatter_kernel<<<(int)(((long)E * D_H + B - 1) / B), B, 0, stream>>>(
        hB, ns, row, col, agg, (long)E * D_H);
    dense128_wmma<<<gDense, 128, 0, stream>>>(agg, nd, WpG2, gcn_b2, hB, 1, N, hA);

    // final features -> f16 for the edge MLP gather
    f32_to_f16<<<(int)((nf + B - 1) / B), B, 0, stream>>>(hA, h16, nf);

    // node scores
    node_mlp<<<(N + 127) / 128, 128, 0, stream>>>(hA, nW0, nb0, nW1, nb1, nW2, nb2,
                                                  node_out, N);
    // link scores (dominant GEMMs)
    edge_mlp_wmma<<<(E + 127) / 128, 256, 0, stream>>>(h16, row, col,
                                                       WpM0, mlp_b0, WpM1, mlp_b1,
                                                       mlp_W2, mlp_b2, link_out, E);
}
